// BioMoR_RNN_6176162972359
// MI455X (gfx1250) — matrix operand order repaired and checked by reading, not compiled, not run on hardware
//
#include <hip/hip_runtime.h>

// Problem constants (from the reference)
constexpr int kH  = 256;     // hidden
constexpr int kB  = 128;     // batch
constexpr int kS  = 2048;    // sequence
constexpr int kG  = 3 * kH;  // 768 gate rows

// B-operand K-chunk split: kc < kRegKc in VGPRs, kc >= kRegKc streamed from LDS
constexpr int kRegKc = 3;
constexpr int kLdsKc = 8 - kRegKc;   // 5

typedef __bf16 bf16_t;
typedef bf16_t v16bf __attribute__((ext_vector_type(16)));
typedef float  v8f   __attribute__((ext_vector_type(8)));

__device__ __forceinline__ float sigmoidf_(float x) {
    return 1.0f / (1.0f + __expf(-x));
}

__device__ __forceinline__ float wredsum(float v) {
#pragma unroll
    for (int off = 16; off >= 1; off >>= 1)
        v += __shfl_xor(v, off, 32);
    return v;
}

__global__ __launch_bounds__(512)
void biomor_rnn_kernel(const float* __restrict__ x,
                       const float* __restrict__ w_ih,
                       const float* __restrict__ w_hh,
                       const float* __restrict__ b_ih,
                       const float* __restrict__ b_hh,
                       const float* __restrict__ w_gain,
                       const float* __restrict__ b_gain,
                       const float* __restrict__ w_bias,
                       const float* __restrict__ b_bias,
                       const float* __restrict__ w_reflex,
                       const float* __restrict__ b_reflex,
                       const float* __restrict__ w_policy,
                       const float* __restrict__ b_policy,
                       const float* __restrict__ w_motor,
                       const float* __restrict__ b_motor,
                       float* __restrict__ out_y,
                       float* __restrict__ out_h)
{
    // ---- LDS: ~301.5 KB total (CDNA5 320 KB WGP LDS pool) ----
    // w_hh B-operand, K-chunks kRegKc..7, pre-packed in WMMA B layout.
    // slot = (wave*3 + tile)*kLdsKc + (kc - kRegKc); slot = 32 lanes * 16 bf16.
    __shared__ __align__(16) bf16_t s_whh2[16 * 3 * kLdsKc * 512];   // 240 KB
    __shared__ __align__(16) bf16_t sh_bf[16 * kH];                  // h as WMMA A source (8 KB)
    __shared__ float sh_f[16 * kH];                                  // h f32 for head dots (16 KB)
    __shared__ float s_wih[kG * 6];                                  // 18 KB
    __shared__ float s_bih[kG];
    __shared__ float s_bhh[kG];
    __shared__ float s_x[16 * 8];
    __shared__ float s_gain[2 * kH];
    __shared__ float s_bias[4 * kH];
    __shared__ float s_reflex[kH * 2];
    __shared__ float s_breflex[kH];
    __shared__ float s_pol[2 * kH];
    __shared__ float s_mot[2 * kH];

    const int tid = threadIdx.x;
    const int wv  = tid >> 5;    // wave 0..15
    const int l   = tid & 31;    // lane
    const int hl  = l >> 4;      // lane half (0/1)
    const int ln  = l & 15;
    const int m0  = blockIdx.x * 16;   // batch-row base for this workgroup

    // ---- cooperative preamble: weights to LDS, h := 0 ----
    for (int i = tid; i < kG * 6; i += 512) s_wih[i] = w_ih[i];
    for (int i = tid; i < kG; i += 512) { s_bih[i] = b_ih[i]; s_bhh[i] = b_hh[i]; }
    for (int i = tid; i < 2 * kH; i += 512) {
        s_gain[i]   = w_gain[i];
        s_pol[i]    = w_policy[i];
        s_mot[i]    = w_motor[i];
        s_reflex[i] = w_reflex[i];   // (H,2) row-major, same flat layout
    }
    for (int i = tid; i < 4 * kH; i += 512) s_bias[i] = w_bias[i];
    for (int i = tid; i < kH; i += 512) s_breflex[i] = b_reflex[i];
    for (int i = tid; i < 16 * kH; i += 512) { sh_f[i] = 0.0f; sh_bf[i] = (bf16_t)0.0f; }

    // x tile for step 0: wave wv owns batch row wv
    if (l < 6) s_x[wv * 8 + l] = x[(((size_t)(m0 + wv)) * kS) * 6 + l];

    // loop-invariant head biases into registers
    const float bg0 = b_gain[0], bg1 = b_gain[1];
    const float bb0 = b_bias[0], bb1 = b_bias[1], bb2 = b_bias[2], bb3 = b_bias[3];
    const float bp0 = b_policy[0], bp1 = b_policy[1];
    const float bm0 = b_motor[0],  bm1 = b_motor[1];

    // ---- w_hh B-operand: wave wv owns gate rows {wv, wv+16, wv+32}*16 + ln ----
    // B as packed here (32x16 bf16): lane ln = N, element i -> K = hl*16 + i (contiguous, as A).
    // K-chunks 0..kRegKc-1 stay in VGPRs (72 regs); rest packed into LDS.
    v16bf Breg[3][kRegKc];
#pragma unroll
    for (int t = 0; t < 3; ++t) {
        const int n = (wv + 16 * t) * 16 + ln;        // gate row index 0..767
        const float* wr = w_hh + (size_t)n * kH;
#pragma unroll
        for (int kc = 0; kc < 8; ++kc) {
            const float* pk = wr + kc * 32 + hl * 16;
            v16bf b;
#pragma unroll
            for (int q = 0; q < 16; ++q) b[q] = (bf16_t)pk[q];
            if (kc < kRegKc) {
                Breg[t][kc] = b;
            } else {
                bf16_t* dst = &s_whh2[(((wv * 3 + t) * kLdsKc + (kc - kRegKc)) * 32 + l) * 16];
                ((float4*)dst)[0] = ((float4*)&b)[0];
                ((float4*)dst)[1] = ((float4*)&b)[1];
            }
        }
    }

    // h_old kept in registers: element r -> (M = r + 8*hl, col c)
    float hold[8];
#pragma unroll
    for (int r = 0; r < 8; ++r) hold[r] = 0.0f;

    const int c = wv * 16 + ln;   // column within each gate block (0..255)

    for (int s = 0; s < kS; ++s) {
        __syncthreads();   // bar1: h(bf16) + x tile + (first iter) s_whh2 visible

        // ---- gh = h @ w_hh^T for this wave's 3 gate tiles ----
        v8f acc0 = {}, acc1 = {}, acc2 = {};
#pragma unroll
        for (int kc = 0; kc < 8; ++kc) {
            // A layout (16x32 bf16): lane ln = M; elem i -> K = kc*32+hl*16+i.
            // Row-major bf16 in LDS => one contiguous 32B run.
            v16bf a;
            const float4* pa = (const float4*)(&sh_bf[ln * kH + kc * 32 + hl * 16]);
            ((float4*)&a)[0] = pa[0];
            ((float4*)&a)[1] = pa[1];   // +16 bytes = +8 elements

            v16bf b0, b1, b2;
            if (kc < kRegKc) {
                b0 = Breg[0][kc]; b1 = Breg[1][kc]; b2 = Breg[2][kc];
            } else {
                const float4* p0 = (const float4*)(&s_whh2[(((wv * 3 + 0) * kLdsKc + (kc - kRegKc)) * 32 + l) * 16]);
                const float4* p1 = (const float4*)(&s_whh2[(((wv * 3 + 1) * kLdsKc + (kc - kRegKc)) * 32 + l) * 16]);
                const float4* p2 = (const float4*)(&s_whh2[(((wv * 3 + 2) * kLdsKc + (kc - kRegKc)) * 32 + l) * 16]);
                ((float4*)&b0)[0] = p0[0]; ((float4*)&b0)[1] = p0[1];
                ((float4*)&b1)[0] = p1[0]; ((float4*)&b1)[1] = p1[1];
                ((float4*)&b2)[0] = p2[0]; ((float4*)&b2)[1] = p2[1];
            }
            acc0 = __builtin_amdgcn_wmma_f32_16x16x32_bf16(false, a, false, b0, (short)0, acc0, false, false);
            acc1 = __builtin_amdgcn_wmma_f32_16x16x32_bf16(false, a, false, b1, (short)0, acc1, false, false);
            acc2 = __builtin_amdgcn_wmma_f32_16x16x32_bf16(false, a, false, b2, (short)0, acc2, false, false);
        }

        // ---- GRU elementwise, fully in-register on accumulator layout ----
#pragma unroll
        for (int r = 0; r < 8; ++r) {
            const int m = r + 8 * hl;              // local batch row 0..15
            const float* xm = &s_x[m * 8];
            const float* wr_ = &s_wih[(size_t)c * 6];
            const float* wz_ = &s_wih[(size_t)(kH + c) * 6];
            const float* wn_ = &s_wih[(size_t)(2 * kH + c) * 6];
            float gir = s_bih[c], giz = s_bih[kH + c], gin = s_bih[2 * kH + c];
#pragma unroll
            for (int d = 0; d < 6; ++d) {
                gir = fmaf(xm[d], wr_[d], gir);
                giz = fmaf(xm[d], wz_[d], giz);
                gin = fmaf(xm[d], wn_[d], gin);
            }
            const float ghr = acc0[r] + s_bhh[c];
            const float ghz = acc1[r] + s_bhh[kH + c];
            const float ghn = acc2[r] + s_bhh[2 * kH + c];
            const float rg = sigmoidf_(gir + ghr);
            const float zg = sigmoidf_(giz + ghz);
            const float ng = tanhf(gin + rg * ghn);
            hold[r] = (1.0f - zg) * ng + zg * hold[r];
        }

        __syncthreads();   // bar2: everyone done reading previous h from LDS

#pragma unroll
        for (int r = 0; r < 8; ++r) {
            const int m = r + 8 * hl;
            sh_f[m * kH + c]  = hold[r];
            sh_bf[m * kH + c] = (bf16_t)hold[r];
            out_h[(((size_t)(m0 + m)) * kS + s) * kH + c] = hold[r];
        }

        __syncthreads();   // bar3: new h visible for head networks

        // ---- heads: wave wv computes batch row wv, wave32 butterfly reductions ----
        {
            const int row = wv;
            float sg0 = 0, sg1 = 0, sb0 = 0, sb1 = 0, sb2 = 0, sb3 = 0;
#pragma unroll
            for (int t = 0; t < 8; ++t) {
                const int k = l + 32 * t;
                const float hk = sh_f[row * kH + k];
                sg0 = fmaf(hk, s_gain[k],          sg0);
                sg1 = fmaf(hk, s_gain[kH + k],     sg1);
                sb0 = fmaf(hk, s_bias[k],          sb0);
                sb1 = fmaf(hk, s_bias[kH + k],     sb1);
                sb2 = fmaf(hk, s_bias[2 * kH + k], sb2);
                sb3 = fmaf(hk, s_bias[3 * kH + k], sb3);
            }
            sg0 = wredsum(sg0); sg1 = wredsum(sg1);
            sb0 = wredsum(sb0); sb1 = wredsum(sb1);
            sb2 = wredsum(sb2); sb3 = wredsum(sb3);
            const float g0 = sigmoidf_(sg0 + bg0);
            const float g1 = sigmoidf_(sg1 + bg1);
            const float a0 = s_x[row * 8 + 0] * g0;
            const float a1 = s_x[row * 8 + 1] * g1;
            float sp0 = 0, sp1 = 0, sm0 = 0, sm1 = 0;
#pragma unroll
            for (int t = 0; t < 8; ++t) {
                const int k = l + 32 * t;
                float f = fmaf(a0, s_reflex[2 * k], fmaf(a1, s_reflex[2 * k + 1], s_breflex[k]));
                f = fmaxf(f, 0.0f);
                sp0 = fmaf(f, s_pol[k],      sp0);
                sp1 = fmaf(f, s_pol[kH + k], sp1);
                sm0 = fmaf(f, s_mot[k],      sm0);
                sm1 = fmaf(f, s_mot[kH + k], sm1);
            }
            sp0 = wredsum(sp0); sp1 = wredsum(sp1);
            sm0 = wredsum(sm0); sm1 = wredsum(sm1);
            if (l == 0) {
                float4 yv;
                yv.x = sp0 + bp0 + (sb0 + bb0);
                yv.y = sp1 + bp1 + (sb1 + bb1);
                yv.z = tanhf(sm0 + bm0 + (sb2 + bb2));
                yv.w = tanhf(sm1 + bm1 + (sb3 + bb3));
                *(float4*)&out_y[(((size_t)(m0 + row)) * kS + s) * 4] = yv;
            }
            // stage x tile for step s+1 (consumed only after next bar1)
            if (s + 1 < kS && l < 6)
                s_x[row * 8 + l] = x[(((size_t)(m0 + row)) * kS + (s + 1)) * 6 + l];
        }
    }
}

extern "C" void kernel_launch(void* const* d_in, const int* in_sizes, int n_in,
                              void* d_out, int out_size, void* d_ws, size_t ws_size,
                              hipStream_t stream) {
    const float* x        = (const float*)d_in[0];
    const float* w_ih     = (const float*)d_in[1];
    const float* w_hh     = (const float*)d_in[2];
    const float* b_ih     = (const float*)d_in[3];
    const float* b_hh     = (const float*)d_in[4];
    const float* w_gain   = (const float*)d_in[5];
    const float* b_gain   = (const float*)d_in[6];
    const float* w_bias   = (const float*)d_in[7];
    const float* b_bias   = (const float*)d_in[8];
    const float* w_reflex = (const float*)d_in[9];
    const float* b_reflex = (const float*)d_in[10];
    const float* w_policy = (const float*)d_in[11];
    const float* b_policy = (const float*)d_in[12];
    const float* w_motor  = (const float*)d_in[13];
    const float* b_motor  = (const float*)d_in[14];

    float* out_y = (float*)d_out;                          // (B,S,4)
    float* out_h = out_y + (size_t)kB * kS * 4;            // (B,S,H)

    biomor_rnn_kernel<<<dim3(kB / 16), dim3(512), 0, stream>>>(
        x, w_ih, w_hh, b_ih, b_hh,
        w_gain, b_gain, w_bias, b_bias,
        w_reflex, b_reflex, w_policy, b_policy, w_motor, b_motor,
        out_y, out_h);
}
